// STU_2302102470914
// MI455X (gfx1250) — matrix operand order repaired and checked by reading, not compiled, not run on hardware
//
#include <hip/hip_runtime.h>
#include <hip/hip_bf16.h>

// ---------------------------------------------------------------------------
// STU forward for MI455X (gfx1250, wave32, WMMA + async LDS copies)
//   b=2, sl=2048, d=768, K=16 filters, k_u=3, k_y=2, padded_sl=4096
// ---------------------------------------------------------------------------

#define SL   2048
#define D    768
#define KF   16
#define NFFT 4096
#define BSZ  2

typedef __attribute__((ext_vector_type(16))) __bf16 bf16x16;
typedef __attribute__((ext_vector_type(8)))  float  v8f;

union FragBF { bf16x16 v; uint4 q[2]; };

__device__ __forceinline__ unsigned short f2bf(float f) {
  unsigned int u = __float_as_uint(f);
  u += 0x7FFFu + ((u >> 16) & 1u);          // round-to-nearest-even
  return (unsigned short)(u >> 16);
}

// ---------------- 4096-point Stockham FFT in LDS (256 threads) --------------
// Data starts in (xr,xi); after 12 stages (even #swaps) result is back in (xr,xi),
// in natural order. sgn = -1 forward, +1 inverse (unscaled).
__device__ void fft4096(float* xr, float* xi, float* yr, float* yi,
                        int tid, float sgn) {
  float* ar = xr; float* ai = xi; float* br = yr; float* bi = yi;
  int n = NFFT, s = 1;
  const float twopi = 6.283185307179586f;
  while (n > 1) {
    int h = n >> 1;
    __syncthreads();
    for (int t = tid; t < (NFFT / 2); t += 256) {
      int i = t & (s - 1);
      int j = t / s;                         // 0..h-1  (s*h == 2048)
      float wi_s, wr_c;
      __sincosf(sgn * twopi * (float)j / (float)n, &wi_s, &wr_c);
      int ia = i + s * j;
      int ib = i + s * (j + h);
      float xar = ar[ia], xai = ai[ia];
      float xbr = ar[ib], xbi = ai[ib];
      float dr = xar - xbr, di = xai - xbi;
      int oa = i + s * (2 * j);
      int ob = i + s * (2 * j + 1);
      br[oa] = xar + xbr;
      bi[oa] = xai + xbi;
      br[ob] = dr * wr_c - di * wi_s;
      bi[ob] = dr * wi_s + di * wr_c;
    }
    float* t1 = ar; ar = br; br = t1;
    float* t2 = ai; ai = bi; bi = t2;
    n = h; s <<= 1;
  }
  __syncthreads();
}

// ---------------- Filter FFTs: V_f and half-spectrum-rotated Vm_f -----------
__global__ __launch_bounds__(256) void stu_fft_filters(
    const float* __restrict__ phi, float2* __restrict__ Vf,
    float2* __restrict__ Vm) {
  __shared__ float xr[NFFT], xi[NFFT], yr[NFFT], yi[NFFT];
  int k = blockIdx.x, tid = threadIdx.x;
  for (int l = tid; l < NFFT; l += 256) {
    xr[l] = (l < SL) ? phi[(size_t)l * KF + k] : 0.f;
    xi[l] = 0.f;
  }
  fft4096(xr, xi, yr, yi, tid, -1.f);
  for (int f = tid; f < NFFT; f += 256) {
    float2 v = make_float2(xr[f], xi[f]);
    Vf[(size_t)k * NFFT + f] = v;
    Vm[(size_t)k * NFFT + ((f + NFFT / 2) & (NFFT - 1))] = v;  // freq shift
  }
}

// ---------------- Forward FFT of every input channel ------------------------
__global__ __launch_bounds__(256) void stu_fft_inputs(
    const float* __restrict__ u, float2* __restrict__ Uf) {
  __shared__ float xr[NFFT], xi[NFFT], yr[NFFT], yi[NFFT];
  int i = blockIdx.x, b = blockIdx.y, tid = threadIdx.x;
  for (int l = tid; l < NFFT; l += 256) {
    xr[l] = (l < SL) ? u[((size_t)b * SL + l) * D + i] : 0.f;
    xi[l] = 0.f;
  }
  fft4096(xr, xi, yr, yi, tid, -1.f);
  float2* dst = Uf + ((size_t)(b * D + i)) * NFFT;
  for (int f = tid; f < NFFT; f += 256) dst[f] = make_float2(xr[f], xi[f]);
}

// ---------------- Per-filter conv (plus + minus) -> bf16 A matrix -----------
// A[row = b*SL + l][col = sign*D + i], row stride 2*D
__global__ __launch_bounds__(256) void stu_conv_k(
    const float2* __restrict__ Uf, const float2* __restrict__ Vf,
    const float2* __restrict__ Vm, unsigned short* __restrict__ A, int k) {
  __shared__ float xr[NFFT], xi[NFFT], yr[NFFT], yi[NFFT];
  int i = blockIdx.x, b = blockIdx.y, tid = threadIdx.x;
  const float2* uf = Uf + ((size_t)(b * D + i)) * NFFT;
  const float inv = 1.f / (float)NFFT;
  for (int sgn = 0; sgn < 2; ++sgn) {
    const float2* vf = (sgn == 0 ? Vf : Vm) + (size_t)k * NFFT;
    __syncthreads();
    for (int f = tid; f < NFFT; f += 256) {
      float2 a = uf[f];
      float2 v = vf[f];
      xr[f] = a.x * v.x - a.y * v.y;
      xi[f] = a.x * v.y + a.y * v.x;
    }
    fft4096(xr, xi, yr, yi, tid, 1.f);
    for (int l = tid; l < SL; l += 256)
      A[((size_t)b * SL + l) * (2 * D) + sgn * D + i] = f2bf(xr[l] * inv);
  }
}

// ---------------- Weight packing (bf16, pre-transposed [N][K]) --------------
__global__ __launch_bounds__(256) void stu_pack_wphi(
    const float* __restrict__ sigma, const float* __restrict__ Mp,
    const float* __restrict__ Mm, unsigned short* __restrict__ Wt) {
  const long long total = (long long)KF * D * (2 * D);   // [k][o][sign*D+i]
  for (long long idx = (long long)blockIdx.x * 256 + threadIdx.x; idx < total;
       idx += (long long)gridDim.x * 256) {
    int k = (int)(idx / ((long long)D * 2 * D));
    long long r = idx % ((long long)D * 2 * D);
    int o = (int)(r / (2 * D));
    int c = (int)(r % (2 * D));
    int sgn = c / D, i = c % D;
    float s = __powf(sigma[k], 0.25f);
    const float* M = (sgn == 0) ? Mp : Mm;
    float m = M[((size_t)k * D + o) * D + i];
    Wt[idx] = f2bf(s * m);
  }
}

__global__ __launch_bounds__(256) void stu_pack_wu(
    const float* __restrict__ Mu, unsigned short* __restrict__ Wut) {
  const long long total = (long long)D * (3 * D);         // [o][i0*D+i]
  for (long long idx = (long long)blockIdx.x * 256 + threadIdx.x; idx < total;
       idx += (long long)gridDim.x * 256) {
    int o = (int)(idx / (3 * D));
    int c = (int)(idx % (3 * D));
    int i0 = c / D, i = c % D;
    Wut[idx] = f2bf(Mu[((size_t)i0 * D + o) * D + i]);
  }
}

__global__ __launch_bounds__(256) void stu_zero(float* __restrict__ p, long long n) {
  for (long long idx = (long long)blockIdx.x * 256 + threadIdx.x; idx < n;
       idx += (long long)gridDim.x * 256) p[idx] = 0.f;
}

// A_u[row][i0*D+i] = u[b, l-i0, i]  (bf16, zero before sequence start)
__global__ __launch_bounds__(256) void stu_build_au(
    const float* __restrict__ u, unsigned short* __restrict__ Au) {
  const long long total = (long long)BSZ * SL * (3 * D);
  for (long long idx = (long long)blockIdx.x * 256 + threadIdx.x; idx < total;
       idx += (long long)gridDim.x * 256) {
    long long row = idx / (3 * D);
    int c = (int)(idx % (3 * D));
    int b = (int)(row >> 11), l = (int)(row & (SL - 1));
    int i0 = c / D, i = c % D;
    int ls = l - i0;
    float v = (ls >= 0) ? u[((size_t)b * SL + ls) * D + i] : 0.f;
    Au[idx] = f2bf(v);
  }
}

// deltas[b,l,o] = spectral[b,l-2,o]  (causal shift by k_y=2)
__global__ __launch_bounds__(256) void stu_build_deltas(
    const float* __restrict__ spec, float* __restrict__ delt) {
  const long long total = (long long)BSZ * SL * D;
  for (long long idx = (long long)blockIdx.x * 256 + threadIdx.x; idx < total;
       idx += (long long)gridDim.x * 256) {
    long long row = idx / D;
    int o = (int)(idx % D);
    int b = (int)(row >> 11), l = (int)(row & (SL - 1));
    delt[idx] = (l >= 2) ? spec[((size_t)b * SL + (l - 2)) * D + o] : 0.f;
  }
}

// ---------------- bf16 WMMA GEMM: C[M][N] += A[M][K] * Bt[N][K]^T -----------
// 256 threads = 8 waves; block tile 128x128, BK=32; each wave: 4x2 16x16 tiles.
// Tile staging uses CDNA5 async LDS copies (GLOBAL_LOAD_ASYNC_TO_LDS_B128,
// tracked by ASYNCcnt) instead of VGPR-staged load/store.
__global__ __launch_bounds__(256) void stu_gemm_bf16(
    const unsigned short* __restrict__ A, const unsigned short* __restrict__ Bt,
    float* __restrict__ C, int M, int N, int K) {
  __shared__ unsigned short As[128][32];
  __shared__ unsigned short Bs[128][32];
  int tid = threadIdx.x;
  int lane = tid & 31;
  int w  = tid >> 5;
  int wm = w & 1;            // M group of 64
  int wn = w >> 1;           // N group of 32
  int m0 = blockIdx.x * 128;
  int n0 = blockIdx.y * 128;

  v8f acc[4][2];
  v8f zero = {0.f, 0.f, 0.f, 0.f, 0.f, 0.f, 0.f, 0.f};
#pragma unroll
  for (int mi = 0; mi < 4; ++mi)
#pragma unroll
    for (int ni = 0; ni < 2; ++ni) acc[mi][ni] = zero;

  int ldr = tid >> 1;          // 0..127: tile row
  int ldc = (tid & 1) * 16;    // 0 or 16: 16 ushorts = 32B per thread
  int lm = lane & 15;
  int hi = lane >> 4;

  // per-lane LDS byte offsets (low 32 bits of the flat address == LDS offset)
  unsigned ldsA = (unsigned)(unsigned long long)&As[ldr][ldc];
  unsigned ldsB = (unsigned)(unsigned long long)&Bs[ldr][ldc];

  for (int kk = 0; kk < K; kk += 32) {
    __syncthreads();   // previous iteration's fragment reads complete
    {
      unsigned long long ga =
          (unsigned long long)(A + (size_t)(m0 + ldr) * K + kk + ldc);
      unsigned long long gb =
          (unsigned long long)(Bt + (size_t)(n0 + ldr) * K + kk + ldc);
      asm volatile(
          "global_load_async_to_lds_b128 %0, %2, off\n\t"
          "global_load_async_to_lds_b128 %0, %2, off offset:16\n\t"
          "global_load_async_to_lds_b128 %1, %3, off\n\t"
          "global_load_async_to_lds_b128 %1, %3, off offset:16"
          :: "v"(ldsA), "v"(ldsB), "v"(ga), "v"(gb)
          : "memory");
    }
    if (kk + 32 < K) {  // hint next K-tile into cache (global_prefetch_b8)
      __builtin_prefetch(A + (size_t)(m0 + ldr) * K + kk + 32 + ldc, 0, 1);
      __builtin_prefetch(Bt + (size_t)(n0 + ldr) * K + kk + 32 + ldc, 0, 1);
    }
    asm volatile("s_wait_asynccnt 0x0" ::: "memory");
    __syncthreads();

    FragBF af[4], bf[2];
#pragma unroll
    for (int mi = 0; mi < 4; ++mi) {
      // A 16x32 bf16 layout: lane m=lane&15; K pairs {hi*8..+7, hi*8+16..+23}
      const uint4* p = (const uint4*)&As[wm * 64 + mi * 16 + lm][0];
      af[mi].q[0] = p[hi];
      af[mi].q[1] = p[hi + 2];
    }
#pragma unroll
    for (int ni = 0; ni < 2; ++ni) {
      // B 32x16 bf16 layout: lane n=lane&15; K pairs {hi*8..+7, hi*8+16..+23}
      const uint4* p = (const uint4*)&Bs[wn * 32 + ni * 16 + lm][0];
      bf[ni].q[0] = p[hi];
      bf[ni].q[1] = p[hi + 2];
    }
#pragma unroll
    for (int mi = 0; mi < 4; ++mi)
#pragma unroll
      for (int ni = 0; ni < 2; ++ni)
        acc[mi][ni] = __builtin_amdgcn_wmma_f32_16x16x32_bf16(
            false, af[mi].v, false, bf[ni].v, (short)0, acc[mi][ni],
            false, false);
  }

  // epilogue: C += acc   (C/D layout: vgpr j -> m = j + 8*hi, lane -> n)
#pragma unroll
  for (int mi = 0; mi < 4; ++mi)
#pragma unroll
    for (int ni = 0; ni < 2; ++ni)
#pragma unroll
      for (int j = 0; j < 8; ++j) {
        int row = m0 + wm * 64 + mi * 16 + j + hi * 8;
        int col = n0 + wn * 32 + ni * 16 + lm;
        C[(size_t)row * N + col] += acc[mi][ni][j];
      }
}

// ---------------- Sequential 2-lag AR recurrence on outputs -----------------
// y_t = delta_t + My[:,0,:] @ y_{t-1} + My[:,1,:] @ y_{t-2}
__global__ __launch_bounds__(768) void stu_ar_y(
    const float* __restrict__ delt, const float* __restrict__ My,
    float* __restrict__ out) {
  __shared__ float y1[D], y2[D];
  int b = blockIdx.x, o = threadIdx.x;
  y1[o] = 0.f; y2[o] = 0.f;
  __syncthreads();
  const float* m0 = My + (size_t)o * (2 * D);
  const float* m1 = m0 + D;
  for (int t = 0; t < SL; ++t) {
    float c = 0.f;
    for (int i = 0; i < D; ++i)
      c = fmaf(m0[i], y1[i], fmaf(m1[i], y2[i], c));
    float yt = delt[((size_t)b * SL + t) * D + o] + c;
    out[((size_t)b * SL + t) * D + o] = yt;
    __syncthreads();
    y2[o] = y1[o];
    y1[o] = yt;
    __syncthreads();
  }
}

// ---------------------------------------------------------------------------
extern "C" void kernel_launch(void* const* d_in, const int* in_sizes, int n_in,
                              void* d_out, int out_size, void* d_ws,
                              size_t ws_size, hipStream_t stream) {
  const float* u     = (const float*)d_in[0];
  const float* sigma = (const float*)d_in[1];
  const float* phi   = (const float*)d_in[2];
  const float* Mu    = (const float*)d_in[3];
  const float* Mp    = (const float*)d_in[4];
  const float* Mm    = (const float*)d_in[5];
  const float* My    = (const float*)d_in[6];
  float* out = (float*)d_out;
  (void)in_sizes; (void)n_in; (void)out_size; (void)ws_size;

  char* ws = (char*)d_ws;
  size_t off = 0;
  auto carve = [&](size_t bytes) -> void* {
    void* p = (void*)(ws + off);
    off += (bytes + 255) & ~(size_t)255;
    return p;
  };
  float2* Vf = (float2*)carve((size_t)KF * NFFT * sizeof(float2));
  float2* Vm = (float2*)carve((size_t)KF * NFFT * sizeof(float2));
  float2* Uf = (float2*)carve((size_t)BSZ * D * NFFT * sizeof(float2));
  unsigned short* Ac  = (unsigned short*)carve((size_t)BSZ * SL * 2 * D * 2);
  unsigned short* Wt  = (unsigned short*)carve((size_t)KF * D * 2 * D * 2);
  float* spec = (float*)carve((size_t)BSZ * SL * D * 4);
  float* delt = (float*)carve((size_t)BSZ * SL * D * 4);
  unsigned short* Au  = (unsigned short*)carve((size_t)BSZ * SL * 3 * D * 2);
  unsigned short* Wut = (unsigned short*)carve((size_t)D * 3 * D * 2);

  // 1) FFTs of filters and inputs
  stu_fft_filters<<<KF, 256, 0, stream>>>(phi, Vf, Vm);
  stu_fft_inputs<<<dim3(D, BSZ), 256, 0, stream>>>(u, Uf);

  // 2) weight packing + zero/init buffers
  stu_pack_wphi<<<2048, 256, 0, stream>>>(sigma, Mp, Mm, Wt);
  stu_pack_wu<<<512, 256, 0, stream>>>(Mu, Wut);
  stu_zero<<<1024, 256, 0, stream>>>(spec, (long long)BSZ * SL * D);
  stu_build_au<<<2048, 256, 0, stream>>>(u, Au);

  // 3) per-filter conv + WMMA GEMM accumulation into spectral
  for (int k = 0; k < KF; ++k) {
    stu_conv_k<<<dim3(D, BSZ), 256, 0, stream>>>(Uf, Vf, Vm, Ac, k);
    stu_gemm_bf16<<<dim3((BSZ * SL) / 128, D / 128), 256, 0, stream>>>(
        Ac, Wt + (size_t)k * D * 2 * D, spec, BSZ * SL, D, 2 * D);
  }

  // 4) deltas = shift(spectral, k_y=2) + A_u @ W_u
  stu_build_deltas<<<1024, 256, 0, stream>>>(spec, delt);
  stu_gemm_bf16<<<dim3((BSZ * SL) / 128, D / 128), 256, 0, stream>>>(
      Au, Wut, delt, BSZ * SL, D, 3 * D);

  // 5) sequential AR over outputs
  stu_ar_y<<<BSZ, D, 0, stream>>>(delt, My, out);
}
